// LSTMmodel_35871566856646
// MI455X (gfx1250) — matrix operand order repaired and checked by reading, not compiled, not run on hardware
//
#include <hip/hip_runtime.h>

// LSTM: x[2048,512,32], U=25 (padded to 32), fused x-proj + recurrence + head.
// One wave32 per 16-row batch tile, one wave per block (128 blocks -> 128 WGPs).
// All 16 weight B-fragments register-resident; V_WMMA_F32_16X16X32_F16 matmuls.

#define B_TOT 2048
#define T_LEN 512
#define F_DIM 32
#define U_DIM 25
#define TILES (B_TOT / 16)   // 128

typedef __attribute__((ext_vector_type(16))) _Float16 v16h;
typedef __attribute__((ext_vector_type(8)))  _Float16 v8h;
typedef __attribute__((ext_vector_type(8)))  float    v8f;
typedef __attribute__((ext_vector_type(4)))  float    v4f;

__device__ __forceinline__ float ftanh(float x) {
#if __has_builtin(__builtin_amdgcn_tanhf)
  return __builtin_amdgcn_tanhf(x);            // V_TANH_F32 (CDNA5 TRANS op)
#elif __has_builtin(__builtin_amdgcn_tanh_f32)
  return __builtin_amdgcn_tanh_f32(x);
#else
  float e = __expf(2.0f * x);
  return (e - 1.0f) / (e + 1.0f);
#endif
}
__device__ __forceinline__ float fsig(float x) {
  // sigmoid(x) = 0.5*tanh(0.5x) + 0.5  -> mul + tanh + fma, no divide
  return fmaf(0.5f, ftanh(0.5f * x), 0.5f);
}

// Build a B-matrix (32x16 K-major) f16 fragment for one 16-column tile of W[kmax,25].
// Lane l: column n = (l&15)+16*ntile, element j: K = (j<8 ? 8*half + j : 16 + 8*half + (j-8)), half = l>>4.
__device__ __forceinline__ v16h make_bfrag(const float* __restrict__ W, int kmax,
                                           int ntile, int lane) {
  const int half = lane >> 4;
  const int n    = (lane & 15) + ntile * 16;
  v16h r;
#pragma unroll
  for (int j = 0; j < 16; ++j) {
    const int k = (j < 8) ? (half * 8 + j) : (16 + half * 8 + (j - 8));
    float v = 0.0f;
    if (k < kmax && n < U_DIM) v = W[k * U_DIM + n];
    r[j] = (_Float16)v;
  }
  return r;
}

__global__ __launch_bounds__(32, 1)
void lstm_fused_wmma(const float* __restrict__ x,
                     const float* __restrict__ Wxf, const float* __restrict__ Whf, const float* __restrict__ bf,
                     const float* __restrict__ Wxi, const float* __restrict__ Whi, const float* __restrict__ bi,
                     const float* __restrict__ Wxm, const float* __restrict__ Whm, const float* __restrict__ bm,
                     const float* __restrict__ Wxo, const float* __restrict__ Who, const float* __restrict__ bo,
                     const float* __restrict__ Wout, const float* __restrict__ bout,
                     float* __restrict__ out)
{
  // Per-wave h relayout buffer: row-major [16 rows][32 cols] f16.
  __shared__ __align__(16) _Float16 hbuf[16][32];   // 1 KB

  const int lane = threadIdx.x & 31;
  const int half = lane >> 4;      // 0: lanes 0-15, 1: lanes 16-31
  const int lr   = lane & 15;

  // All weight fragments register-resident (loaded once; survive the loop fence).
  const v16h bxf0 = make_bfrag(Wxf, F_DIM, 0, lane), bxf1 = make_bfrag(Wxf, F_DIM, 1, lane);
  const v16h bxi0 = make_bfrag(Wxi, F_DIM, 0, lane), bxi1 = make_bfrag(Wxi, F_DIM, 1, lane);
  const v16h bxm0 = make_bfrag(Wxm, F_DIM, 0, lane), bxm1 = make_bfrag(Wxm, F_DIM, 1, lane);
  const v16h bxo0 = make_bfrag(Wxo, F_DIM, 0, lane), bxo1 = make_bfrag(Wxo, F_DIM, 1, lane);
  const v16h bhf0 = make_bfrag(Whf, U_DIM, 0, lane), bhf1 = make_bfrag(Whf, U_DIM, 1, lane);
  const v16h bhi0 = make_bfrag(Whi, U_DIM, 0, lane), bhi1 = make_bfrag(Whi, U_DIM, 1, lane);
  const v16h bhm0 = make_bfrag(Whm, U_DIM, 0, lane), bhm1 = make_bfrag(Whm, U_DIM, 1, lane);
  const v16h bho0 = make_bfrag(Who, U_DIM, 0, lane), bho1 = make_bfrag(Who, U_DIM, 1, lane);

  // Per-lane bias values (C/D layout: lane encodes column).
  const int n0 = lr, n1 = lr + 16;
  const float bf0 = (n0 < U_DIM) ? bf[n0] : 0.0f, bf1 = (n1 < U_DIM) ? bf[n1] : 0.0f;
  const float bi0 = (n0 < U_DIM) ? bi[n0] : 0.0f, bi1 = (n1 < U_DIM) ? bi[n1] : 0.0f;
  const float bm0 = (n0 < U_DIM) ? bm[n0] : 0.0f, bm1 = (n1 < U_DIM) ? bm[n1] : 0.0f;
  const float bo0 = (n0 < U_DIM) ? bo[n0] : 0.0f, bo1 = (n1 < U_DIM) ? bo[n1] : 0.0f;

  const int tile    = blockIdx.x;
  const int rowBase = tile * 16;
  // A-matrix layout: lane owns batch row (lane&15); feature/K groups by lane half.
  const float* xrow = x + ((size_t)(rowBase + lr) * T_LEN) * F_DIM;

  v8f h0 = {}, h1 = {}, c0 = {}, c1 = {};   // 16x32 state in C/D fragments, f32

  for (int t = 0; t < T_LEN; ++t) {
    // ---- stream x_t tile first (longest latency; overlaps the LDS relayout) ----
    const float* xp = xrow + t * F_DIM + half * 8;
    const v4f xa = *(const v4f*)(xp);
    const v4f xb = *(const v4f*)(xp + 4);
    const v4f xc = *(const v4f*)(xp + 16);
    const v4f xd = *(const v4f*)(xp + 20);
    __builtin_prefetch(xp + 8 * F_DIM, 0, 3);   // global_prefetch_b8, ~8 steps ahead

    // ---- relayout h: C/D fragments -> row-major f16 LDS -> A fragment ----
#pragma unroll
    for (int v = 0; v < 8; ++v) {
      hbuf[half * 8 + v][lr]      = (_Float16)h0[v];
      hbuf[half * 8 + v][16 + lr] = (_Float16)h1[v];
    }
    asm volatile("" ::: "memory");   // same-wave LDS ops are in-order; fence the compiler
    const v8h* rp = (const v8h*)&hbuf[lr][0];
    const v8h ah_lo = rp[half];        // K = {0-7 | 8-15}
    const v8h ah_hi = rp[half + 2];    // K = {16-23 | 24-31}
    v16h ah;
#pragma unroll
    for (int j = 0; j < 8; ++j) { ah[j] = ah_lo[j]; ah[8 + j] = ah_hi[j]; }

    v16h ax;
#pragma unroll
    for (int j = 0; j < 4; ++j) {
      ax[j]      = (_Float16)xa[j];
      ax[4 + j]  = (_Float16)xb[j];
      ax[8 + j]  = (_Float16)xc[j];
      ax[12 + j] = (_Float16)xd[j];
    }

    // ---- 4 gates x 2 N-tiles: D = bias + h@Wh + x_t@Wx (f32 accumulate) ----
    auto gate = [&](const v16h& bh, const v16h& bx, float bias) -> v8f {
      v8f acc;
#pragma unroll
      for (int j = 0; j < 8; ++j) acc[j] = bias;
      acc = __builtin_amdgcn_wmma_f32_16x16x32_f16(false, ah, false, bh,
                                                   (short)0, acc, false, false);
      acc = __builtin_amdgcn_wmma_f32_16x16x32_f16(false, ax, false, bx,
                                                   (short)0, acc, false, false);
      return acc;
    };
    const v8f pf0 = gate(bhf0, bxf0, bf0);
    const v8f pf1 = gate(bhf1, bxf1, bf1);
    const v8f pi0 = gate(bhi0, bxi0, bi0);
    const v8f pi1 = gate(bhi1, bxi1, bi1);
    const v8f pm0 = gate(bhm0, bxm0, bm0);
    const v8f pm1 = gate(bhm1, bxm1, bm1);
    const v8f po0 = gate(bho0, bxo0, bo0);
    const v8f po1 = gate(bho1, bxo1, bo1);

    // ---- elementwise LSTM cell update (fragments are elementwise-aligned) ----
#pragma unroll
    for (int j = 0; j < 8; ++j) {
      const float f0 = fsig(pf0[j]), i0 = fsig(pi0[j]);
      const float m0 = ftanh(pm0[j]), o0 = fsig(po0[j]);
      c0[j] = f0 * c0[j] + i0 * m0;
      h0[j] = o0 * ftanh(c0[j]);
      const float f1 = fsig(pf1[j]), i1 = fsig(pi1[j]);
      const float m1 = ftanh(pm1[j]), o1 = fsig(po1[j]);
      c1[j] = f1 * c1[j] + i1 * m1;
      h1[j] = o1 * ftanh(c1[j]);
    }
  }

  // ---- head: out[row] = sigmoid(h . Wout + bout), cross-lane column reduction ----
  const float w0 = (n0 < U_DIM) ? Wout[n0] : 0.0f;
  const float w1 = (n1 < U_DIM) ? Wout[n1] : 0.0f;
  v8f part;
#pragma unroll
  for (int j = 0; j < 8; ++j) part[j] = h0[j] * w0 + h1[j] * w1;
#pragma unroll
  for (int off = 8; off >= 1; off >>= 1) {
#pragma unroll
    for (int j = 0; j < 8; ++j) part[j] += __shfl_xor(part[j], off, 32);
  }
  const float bo_ = bout[0];
  if (lr == 0) {   // lanes 0 (rows 0-7) and 16 (rows 8-15)
#pragma unroll
    for (int j = 0; j < 8; ++j)
      out[rowBase + half * 8 + j] = fsig(part[j] + bo_);
  }
}

extern "C" void kernel_launch(void* const* d_in, const int* in_sizes, int n_in,
                              void* d_out, int out_size, void* d_ws, size_t ws_size,
                              hipStream_t stream) {
  const float* x    = (const float*)d_in[0];
  const float* Wxf  = (const float*)d_in[1];
  const float* Whf  = (const float*)d_in[2];
  const float* bf   = (const float*)d_in[3];
  const float* Wxi  = (const float*)d_in[4];
  const float* Whi  = (const float*)d_in[5];
  const float* bi   = (const float*)d_in[6];
  const float* Wxm  = (const float*)d_in[7];
  const float* Whm  = (const float*)d_in[8];
  const float* bm   = (const float*)d_in[9];
  const float* Wxo  = (const float*)d_in[10];
  const float* Who  = (const float*)d_in[11];
  const float* bo   = (const float*)d_in[12];
  const float* Wout = (const float*)d_in[13];
  const float* bout = (const float*)d_in[14];
  float* out = (float*)d_out;

  dim3 grid(TILES);   // 128 blocks, one wave32 each -> spread across WGPs
  dim3 block(32);
  lstm_fused_wmma<<<grid, block, 0, stream>>>(x,
      Wxf, Whf, bf, Wxi, Whi, bi, Wxm, Whm, bm, Wxo, Who, bo, Wout, bout, out);
}
